// GroupedQueryAttention_61727269978763
// MI455X (gfx1250) — matrix operand order repaired and checked by reading, not compiled, not run on hardware
//
#include <hip/hip_runtime.h>
#include <hip/hip_bf16.h>
#include <math.h>

// ---------------------------------------------------------------------------
// GQA forward for MI455X (gfx1250, wave32, WMMA 16x16x32 bf16 -> f32 acc)
// Pipeline:
//   1) cvt f32->bf16 for x, wq, wk, wv, wo
//   2) gemm16x64<ROPE>  : Q = x@wq^T, fused RoPE, head-major bf16
//   3) gemm16x64<ROPE>  : K = x@wk^T, fused RoPE, head-major bf16
//   4) gemm16x64<VT>    : V = x@wv^T, transposed bf16 [b][kv][hd][s]
//   5) attn_fwd         : flash attention; K/V tiles double-buffered in LDS
//                         via the Tensor Data Mover (tensor_load_to_lds +
//                         in-order s_wait_tensorcnt), WMMA QK^T + PV
//   6) gemm16x64<OUTF32>: out = Acat@wo^T, f32
// ---------------------------------------------------------------------------

typedef __bf16 bf16_t;
typedef __bf16 v16bf __attribute__((ext_vector_type(16)));
typedef __bf16 v8bf  __attribute__((ext_vector_type(8)));
typedef float  v8f   __attribute__((ext_vector_type(8)));

#define S_LEN   2048
#define DMODEL  2048
#define NH      32
#define NKV     8
#define HD      64
#define BATCH   2
#define MROWS   (BATCH * S_LEN)   // 4096
#define AWAVES  4                 // waves per block in attn_fwd

#if defined(__HIP_DEVICE_COMPILE__) && __has_builtin(__builtin_amdgcn_tensor_load_to_lds) && __has_builtin(__builtin_amdgcn_s_wait_tensorcnt)
#define HAVE_TDM 1
#else
#define HAVE_TDM 0
#endif

__device__ __forceinline__ v8f wmma_bf16(v16bf a, v16bf b, v8f c) {
  return __builtin_amdgcn_wmma_f32_16x16x32_bf16(
      /*neg_a=*/false, a, /*neg_b=*/false, b,
      /*c_mod=*/(short)0, c, /*reuse_a=*/false, /*reuse_b=*/false);
}

// A-operand (16x32, 16-bit), contiguous k order (matches load_bT):
//   lane<16 : halves 0..15 -> K=k0..k0+15
//   lane>=16: halves 0..15 -> K=k0+16..+31
__device__ __forceinline__ v16bf load_a_g(const bf16_t* base, int ldk, int m0, int k0) {
  const int lane = threadIdx.x & 31;
  const int m  = m0 + (lane & 15);
  const int kr = k0 + ((lane < 16) ? 0 : 16);
  const bf16_t* p = base + (size_t)m * ldk + kr;
  return *(const v16bf*)(p);
}

// B-operand (32x16, 16-bit), B[k][n] = src[n0+n][k0+k]:
//   lane<16 : halves 0..15 -> K=k0..k0+15   (N = lane)
//   lane>=16: halves 0..15 -> K=k0+16..+31  (N = lane-16)
// One contiguous 32B access per lane (global or LDS).
__device__ __forceinline__ v16bf load_bT(const bf16_t* base, int ldk, int n0, int k0) {
  const int lane = threadIdx.x & 31;
  const int n = n0 + (lane & 15);
  const int k = k0 + ((lane < 16) ? 0 : 16);
  return *(const v16bf*)(base + (size_t)n * ldk + k);
}

#if HAVE_TDM
typedef unsigned int u32x4 __attribute__((ext_vector_type(4)));
typedef int i32x4 __attribute__((ext_vector_type(4)));
typedef int i32x8 __attribute__((ext_vector_type(8)));

// TDM 2D tile load, bf16 elements, per ISA cdna5 D# layout (groups 0/1).
// global_addr = tile start; dims are remaining extents from tile origin.
__device__ __forceinline__ void tdm_load_2d(const void* gsrc, void* lds_dst,
                                            unsigned tensor_w, unsigned tensor_h,
                                            unsigned row_stride,
                                            unsigned tile_w, unsigned tile_h) {
  unsigned long long ga = (unsigned long long)gsrc;
  unsigned lds = (unsigned)(unsigned long long)lds_dst;   // low 32b = LDS offset
  u32x4 g0;
  g0[0] = 1u;                                             // count=1, user desc
  g0[1] = lds;                                            // lds_addr
  g0[2] = (unsigned)(ga & 0xffffffffu);                   // global_addr[31:0]
  g0[3] = (unsigned)((ga >> 32) & 0x01ffffffu) | (2u << 30); // addr[56:32]|type=2
  i32x8 g1;
  g1[0] = 1 << 16;                                        // data_size=1 (2B)
  g1[1] = (int)((tensor_w & 0xffffu) << 16);              // tensor_dim0[15:0]
  g1[2] = (int)(((tensor_w >> 16) & 0xffffu) | ((tensor_h & 0xffffu) << 16));
  g1[3] = (int)(((tensor_h >> 16) & 0xffffu) | ((tile_w & 0xffffu) << 16));
  g1[4] = (int)(tile_h & 0xffffu);                        // tile_dim1, tile_dim2=0
  g1[5] = (int)row_stride;                                // tensor_dim0_stride
  g1[6] = 0;
  g1[7] = 0;
  i32x4 z4 = {0, 0, 0, 0};
#if __clang_major__ >= 23
  i32x8 z8 = {0, 0, 0, 0, 0, 0, 0, 0};
  __builtin_amdgcn_tensor_load_to_lds(g0, g1, z4, z4, z8, 0);
#else
  __builtin_amdgcn_tensor_load_to_lds(g0, g1, z4, z4, 0);
#endif
}
#endif

__global__ void cvt_f32_bf16(const float* __restrict__ src, bf16_t* __restrict__ dst, int n) {
  int i = blockIdx.x * blockDim.x + threadIdx.x;
  if (i < n) dst[i] = (bf16_t)src[i];
}

enum { MODE_ROPE = 0, MODE_VT = 1, MODE_OUTF32 = 2 };

// One wave computes a 16x64 tile of C = A[4096,K] @ W[N,K]^T.
// C-layout element: (M = r + 8*(lane>>4), N = lane&15) in acc[c][r].
template <int MODE>
__global__ __launch_bounds__(256) void gemm16x64(
    const bf16_t* __restrict__ A, const bf16_t* __restrict__ W,
    void* __restrict__ Out, int Ntot, int Ktot) {
  const int wave   = blockIdx.x * (blockDim.x >> 5) + (threadIdx.x >> 5);
  const int tilesN = Ntot >> 6;
  const int tm = wave / tilesN;
  const int tn = wave % tilesN;
  const int lane  = threadIdx.x & 31;
  const int nIdx  = lane & 15;
  const int mhalf = (lane >> 4) << 3;
  const int m0 = tm << 4;
  const int n0 = tn << 6;

  v8f acc[4] = {};
  for (int k0 = 0; k0 < Ktot; k0 += 32) {
    v16bf va = load_a_g(A, Ktot, m0, k0);
#pragma unroll
    for (int c = 0; c < 4; ++c) {
      v16bf vb = load_bT(W, Ktot, n0 + (c << 4), k0);
      acc[c] = wmma_bf16(va, vb, acc[c]);
    }
    if (k0 + 32 < Ktot)  // global_prefetch_b8 the next A slice
      __builtin_prefetch(A + (size_t)(m0 + nIdx) * Ktot + k0 + 32, 0, 3);
  }

#pragma unroll
  for (int c = 0; c < 4; ++c) {
#pragma unroll
    for (int r = 0; r < 8; ++r) {
      float v = acc[c][r];
      const int row = m0 + r + mhalf;
      const int col = n0 + (c << 4) + nIdx;
      if (MODE == MODE_OUTF32) {
        ((float*)Out)[(size_t)row * Ntot + col] = v;
      } else {
        const int b = row / S_LEN, s = row % S_LEN;
        const int h = col >> 6, f = col & 63;
        const int nheads = Ntot >> 6;
        if (MODE == MODE_ROPE) {
          // pair feature (f^1) lives in lane^1, same register
          float partner = __shfl_xor(v, 1, 32);
          float theta = __expf((float)(f >> 1) * (-0.2878231366f)); // -ln(1e4)/32
          float sn, cs;
          __sincosf((float)s * theta, &sn, &cs);
          float o = (f & 1) ? (v * cs + partner * sn) : (v * cs - partner * sn);
          ((bf16_t*)Out)[(((size_t)(b * nheads + h) * S_LEN + s) << 6) + f] = (bf16_t)o;
        } else { // MODE_VT : Vt[b][h][f][s]
          ((bf16_t*)Out)[((size_t)(b * nheads + h) * HD + f) * S_LEN + s] = (bf16_t)v;
        }
      }
    }
  }
}

// Flash attention: one wave per 16-row q-tile of one (b,h).
__global__ __launch_bounds__(AWAVES * 32) void attn_fwd(
    const bf16_t* __restrict__ Qh, const bf16_t* __restrict__ Kh,
    const bf16_t* __restrict__ Vt, bf16_t* __restrict__ Acat) {
  __shared__ __align__(16) bf16_t pbuf[AWAVES][16 * 32];      // P staging
#if HAVE_TDM
  __shared__ __align__(16) bf16_t kbuf[AWAVES][2][32 * HD];   // K tiles (TDM, dbl-buf)
  __shared__ __align__(16) bf16_t vbuf[AWAVES][2][HD * 32];   // V^T tiles (TDM, dbl-buf)
#endif
  const int widx = threadIdx.x >> 5;
  const int wave = blockIdx.x * AWAVES + widx;
  const int qt = wave & 127;          // S/16
  const int h  = (wave >> 7) & 31;
  const int b  = wave >> 12;
  const int kv = h >> 2;              // n_rep = 4
  const int lane  = threadIdx.x & 31;
  const int nIdx  = lane & 15;
  const int mhalf = (lane >> 4) << 3;
  const int qbase = qt << 4;

  const bf16_t* Qp = Qh + (size_t)(b * NH + h) * S_LEN * HD;
  const bf16_t* Kp = Kh + (size_t)(b * NKV + kv) * S_LEN * HD;
  const bf16_t* Vp = Vt + (size_t)(b * NKV + kv) * HD * S_LEN;

  const v16bf qa0 = load_a_g(Qp, HD, qbase, 0);
  const v16bf qa1 = load_a_g(Qp, HD, qbase, 32);

  v8f o[4] = {};
  float mrow[8], lrow[8];
#pragma unroll
  for (int r = 0; r < 8; ++r) { mrow[r] = -1e30f; lrow[r] = 0.f; }

  bf16_t* pb = pbuf[widx];
  const int jmask = qbase >> 5;        // first block that needs causal masking
  const int jmax  = (qbase + 15) >> 5; // inclusive
#if HAVE_TDM
  // Prime the pipeline: block 0 into buffer 0.
  tdm_load_2d(Kp, kbuf[widx][0], HD, S_LEN, HD, HD, 32);
  tdm_load_2d(Vp, vbuf[widx][0], S_LEN, HD, S_LEN, 32, HD);
#endif
  for (int j = 0; j <= jmax; ++j) {
    const int kb = j << 5;
#if HAVE_TDM
    // Prefetch block j+1 into the alternate buffer, then wait for block j's
    // pair (TENSORcnt decrements in order, so <=2 outstanding means the older
    // pair has landed).
    if (j < jmax) {
      const int kb2 = kb + 32;
      tdm_load_2d(Kp + (size_t)kb2 * HD, kbuf[widx][(j + 1) & 1],
                  HD, S_LEN - kb2, HD, HD, 32);
      tdm_load_2d(Vp + kb2, vbuf[widx][(j + 1) & 1],
                  S_LEN - kb2, HD, S_LEN, 32, HD);
      __builtin_amdgcn_s_wait_tensorcnt((short)2);
    } else {
      __builtin_amdgcn_s_wait_tensorcnt((short)0);
    }
    const bf16_t* kl = kbuf[widx][j & 1];
    const bf16_t* vl = vbuf[widx][j & 1];
#endif
    v8f sc[2];
#pragma unroll
    for (int c = 0; c < 2; ++c) {
#if HAVE_TDM
      v16bf kb0 = load_bT(kl, HD, (c << 4), 0);
      v16bf kb1 = load_bT(kl, HD, (c << 4), 32);
#else
      v16bf kb0 = load_bT(Kp, HD, kb + (c << 4), 0);
      v16bf kb1 = load_bT(Kp, HD, kb + (c << 4), 32);
#endif
      v8f z = {};
      z = wmma_bf16(qa0, kb0, z);
      sc[c] = wmma_bf16(qa1, kb1, z);
    }
    // scale (1/sqrt(64)); causal mask only on boundary blocks
    if (j >= jmask) {
#pragma unroll
      for (int c = 0; c < 2; ++c)
#pragma unroll
        for (int r = 0; r < 8; ++r) {
          const int q = qbase + r + mhalf;
          const int k = kb + (c << 4) + nIdx;
          float s = sc[c][r] * 0.125f;
          sc[c][r] = (k > q) ? -1e30f : s;
        }
    } else {
#pragma unroll
      for (int c = 0; c < 2; ++c)
#pragma unroll
        for (int r = 0; r < 8; ++r) sc[c][r] *= 0.125f;
    }
    // online softmax (row = 16 lanes within a half-wave)
#pragma unroll
    for (int r = 0; r < 8; ++r) {
      float mx = fmaxf(sc[0][r], sc[1][r]);
      mx = fmaxf(mx, __shfl_xor(mx, 1, 32));
      mx = fmaxf(mx, __shfl_xor(mx, 2, 32));
      mx = fmaxf(mx, __shfl_xor(mx, 4, 32));
      mx = fmaxf(mx, __shfl_xor(mx, 8, 32));
      const float mn = fmaxf(mrow[r], mx);
      const float al = __expf(mrow[r] - mn);
      float p0 = __expf(sc[0][r] - mn);
      float p1 = __expf(sc[1][r] - mn);
      sc[0][r] = p0; sc[1][r] = p1;
      float rs = p0 + p1;
      rs += __shfl_xor(rs, 1, 32);
      rs += __shfl_xor(rs, 2, 32);
      rs += __shfl_xor(rs, 4, 32);
      rs += __shfl_xor(rs, 8, 32);
      lrow[r] = lrow[r] * al + rs;
      mrow[r] = mn;
#pragma unroll
      for (int c2 = 0; c2 < 4; ++c2) o[c2][r] *= al;
    }
    // C-layout P -> LDS (row-major 16x32 bf16) -> A-operand
#pragma unroll
    for (int c = 0; c < 2; ++c)
#pragma unroll
      for (int r = 0; r < 8; ++r)
        pb[(r + mhalf) * 32 + (c << 4) + nIdx] = (bf16_t)sc[c][r];
    {
      const int kr = (lane < 16) ? 0 : 16;
      const bf16_t* pp = pb + nIdx * 32 + kr;
      v16bf pa = *(const v16bf*)(pp);
#pragma unroll
      for (int c2 = 0; c2 < 4; ++c2) {
#if HAVE_TDM
        v16bf vb = load_bT(vl, 32, c2 << 4, 0);          // B[k][n] = Vtile[feat n][key k]
#else
        v16bf vb = load_bT(Vp, S_LEN, c2 << 4, kb);
#endif
        o[c2] = wmma_bf16(pa, vb, o[c2]);
      }
    }
  }
  // normalize + store concat layout [b*S+q][h*64+f] (bf16 for final GEMM)
#pragma unroll
  for (int c2 = 0; c2 < 4; ++c2)
#pragma unroll
    for (int r = 0; r < 8; ++r) {
      const float v = o[c2][r] / lrow[r];
      const int q = qbase + r + mhalf;
      const int col = (h << 6) + (c2 << 4) + nIdx;
      Acat[((size_t)b * S_LEN + q) * DMODEL + col] = (bf16_t)v;
    }
}

extern "C" void kernel_launch(void* const* d_in, const int* in_sizes, int n_in,
                              void* d_out, int out_size, void* d_ws, size_t ws_size,
                              hipStream_t stream) {
  (void)in_sizes; (void)n_in; (void)out_size; (void)ws_size;
  const float* x  = (const float*)d_in[0];
  const float* wq = (const float*)d_in[1];
  const float* wk = (const float*)d_in[2];
  const float* wv = (const float*)d_in[3];
  const float* wo = (const float*)d_in[4];

  char* ws = (char*)d_ws;
  size_t off = 0;
  auto alloc = [&](size_t elems) {
    bf16_t* p = (bf16_t*)(ws + off);
    off += elems * sizeof(bf16_t);
    return p;
  };
  bf16_t* xb   = alloc((size_t)MROWS * DMODEL);        // 16 MB
  bf16_t* wqb  = alloc((size_t)DMODEL * DMODEL);       //  8 MB
  bf16_t* wkb  = alloc((size_t)(NKV * HD) * DMODEL);   //  2 MB
  bf16_t* wvb  = alloc((size_t)(NKV * HD) * DMODEL);   //  2 MB
  bf16_t* wob  = alloc((size_t)DMODEL * DMODEL);       //  8 MB
  bf16_t* Qhb  = alloc((size_t)BATCH * NH  * S_LEN * HD);  // 16 MB
  bf16_t* Khb  = alloc((size_t)BATCH * NKV * S_LEN * HD);  //  4 MB
  bf16_t* Vtb  = alloc((size_t)BATCH * NKV * HD * S_LEN);  //  4 MB
  bf16_t* Acat = alloc((size_t)MROWS * DMODEL);        // 16 MB

  auto cvt = [&](const float* s, bf16_t* d, size_t n) {
    cvt_f32_bf16<<<(unsigned)((n + 255) / 256), 256, 0, stream>>>(s, d, (int)n);
  };
  cvt(x,  xb,  (size_t)MROWS * DMODEL);
  cvt(wq, wqb, (size_t)DMODEL * DMODEL);
  cvt(wk, wkb, (size_t)(NKV * HD) * DMODEL);
  cvt(wv, wvb, (size_t)(NKV * HD) * DMODEL);
  cvt(wo, wob, (size_t)DMODEL * DMODEL);

  // blocks = (M/16 tiles_m) * (N/64 tiles_n) waves / 8 waves-per-block
  const unsigned blkQ = (MROWS / 16) * (DMODEL / 64) / 8;     // 1024
  const unsigned blkK = (MROWS / 16) * ((NKV * HD) / 64) / 8; // 256
  gemm16x64<MODE_ROPE><<<blkQ, 256, 0, stream>>>(xb, wqb, Qhb, DMODEL, DMODEL);
  gemm16x64<MODE_ROPE><<<blkK, 256, 0, stream>>>(xb, wkb, Khb, NKV * HD, DMODEL);
  gemm16x64<MODE_VT><<<blkK, 256, 0, stream>>>(xb, wvb, Vtb, NKV * HD, DMODEL);

  attn_fwd<<<(BATCH * NH * (S_LEN / 16)) / AWAVES, AWAVES * 32, 0, stream>>>(
      Qhb, Khb, Vtb, Acat);

  gemm16x64<MODE_OUTF32><<<blkQ, 256, 0, stream>>>(Acat, wob, d_out, DMODEL, DMODEL);
}
